// E_GCL_35304631173698
// MI455X (gfx1250) — matrix operand order repaired and checked by reading, not compiled, not run on hardware
//
#include <hip/hip_runtime.h>
#include <hip/hip_bf16.h>
#include <math.h>

#define NODES 50000
#define EDGES 400000
#define NF 320
#define HID 320
#define K_E1 672      // 2*NF+1 = 641 padded up to 21*32
#define K_N1 640      // NF + HID = 20*32
#define EPSV 1e-8f

typedef __attribute__((ext_vector_type(16))) _Float16 v16h;
typedef __attribute__((ext_vector_type(8)))  _Float16 v8h;
typedef __attribute__((ext_vector_type(4)))  _Float16 v4h;
typedef __attribute__((ext_vector_type(8)))  float    v8f;

__device__ __forceinline__ float silu_f(float x) { return x / (1.0f + __expf(-x)); }

// A fragment (16x32 f16) from LDS row-major X[16][lda]:
// lane l -> row l&15 ; halves 0..7 at K=k0+(l>=16)*8+e ; 8..15 at K=k0+16+(l>=16)*8+e
__device__ __forceinline__ v16h load_a_lds(const _Float16* X, int lda, int k0, int lane) {
    int row = lane & 15;
    int kb  = (lane >> 4) * 8;
    const v8h* p0 = reinterpret_cast<const v8h*>(X + row * lda + k0 + kb);
    const v8h* p1 = reinterpret_cast<const v8h*>(X + row * lda + k0 + 16 + kb);
    v8h lo = *p0, hi = *p1;
    v16h r;
#pragma unroll
    for (int i = 0; i < 8; ++i) { r[i] = lo[i]; r[i + 8] = hi[i]; }
    return r;
}

// B fragment (32x16 f16) for Y = X * W^T with W row-major [out][ldw]:
// lane l -> column n0+(l&15); halves 0..7 at K=k0+(l>=16)*8+e ; 8..15 at K=k0+16+(l>=16)*8+e
__device__ __forceinline__ v16h load_b_w(const _Float16* W, int ldw, int n0, int k0, int lane) {
    int n = n0 + (lane & 15);
    int k = k0 + (lane >> 4) * 8;
    v8h lo = *reinterpret_cast<const v8h*>(W + n * ldw + k);
    v8h hi = *reinterpret_cast<const v8h*>(W + n * ldw + k + 16);
    v16h r;
#pragma unroll
    for (int i = 0; i < 8; ++i) { r[i] = lo[i]; r[i + 8] = hi[i]; }
    return r;
}

// Two 16x16 output tiles (rows 0..15 and 16..31 of A) sharing each B fragment.
__device__ __forceinline__ void wmma_2m(const _Float16* A, int lda,
                                        const _Float16* W, int ldw, int K,
                                        int ntile, float biasval, int lane,
                                        v8f& acc0, v8f& acc1) {
#pragma unroll
    for (int i = 0; i < 8; ++i) { acc0[i] = biasval; acc1[i] = biasval; }
    for (int k0 = 0; k0 < K; k0 += 32) {
        v16h b  = load_b_w(W, ldw, ntile * 16, k0, lane);
        v16h a0 = load_a_lds(A,            lda, k0, lane);
        v16h a1 = load_a_lds(A + 16 * lda, lda, k0, lane);
        acc0 = __builtin_amdgcn_wmma_f32_16x16x32_f16(false, a0, false, b,
                                                      (short)0, acc0, false, false);
        acc1 = __builtin_amdgcn_wmma_f32_16x16x32_f16(false, a1, false, b,
                                                      (short)0, acc1, false, false);
    }
}

// ---------------- weight fp32 -> fp16 (with K padding) ----------------
__global__ void cvt_w_kernel(const float* __restrict__ src, _Float16* __restrict__ dst,
                             int rows, int cols, int ldk) {
    int i = blockIdx.x * blockDim.x + threadIdx.x;
    int tot = rows * ldk;
    if (i < tot) {
        int r = i / ldk, c = i - r * ldk;
        dst[i] = (c < cols) ? (_Float16)src[r * cols + c] : (_Float16)0.0f;
    }
}

__global__ void zero_kernel(float* __restrict__ p, int n) {
    int i = blockIdx.x * blockDim.x + threadIdx.x;
    if (i < n) p[i] = 0.0f;
}

// ---------------- edge kernel: 32 edges per block, 8 waves -----------------
__global__ __launch_bounds__(256)
void egcl_edge_kernel(const float* __restrict__ h, const float* __restrict__ coord,
                      const int* __restrict__ eidx,
                      const _Float16* __restrict__ We1, const float* __restrict__ be1,
                      const _Float16* __restrict__ We2, const float* __restrict__ be2,
                      const _Float16* __restrict__ Wc1, const float* __restrict__ bc1,
                      const float* __restrict__ Wc2,
                      float* __restrict__ agg, float* __restrict__ coord_agg) {
    __shared__ __align__(16) _Float16 s_ein[32 * K_E1];   // 43008 B
    __shared__ __align__(16) _Float16 s_m1[32 * HID];     // 20480 B (reused for coord-head c)
    __shared__ __align__(16) _Float16 s_m2[32 * HID];     // 20480 B
    __shared__ int   s_row[32];
    __shared__ int   s_col[32];
    __shared__ float s_cd[32 * 3];

    const int tid  = threadIdx.x;
    const int lane = tid & 31;
    const int wave = tid >> 5;
    const int e0   = blockIdx.x * 32;

    if (tid < 32) {
        int e = tid;
        int r = eidx[e0 + e];
        int c = eidx[EDGES + e0 + e];
        s_row[e] = r; s_col[e] = c;
        float dx = coord[r * 3 + 0] - coord[c * 3 + 0];
        float dy = coord[r * 3 + 1] - coord[c * 3 + 1];
        float dz = coord[r * 3 + 2] - coord[c * 3 + 2];
        float rad = dx * dx + dy * dy + dz * dz;
        float inv = 1.0f / (sqrtf(rad) + EPSV);
        s_cd[e * 3 + 0] = dx * inv;
        s_cd[e * 3 + 1] = dy * inv;
        s_cd[e * 3 + 2] = dz * inv;
        s_ein[e * K_E1 + 640] = (_Float16)rad;
#pragma unroll
        for (int k = 641; k < K_E1; ++k) s_ein[e * K_E1 + k] = (_Float16)0.0f;
    }
    __syncthreads();

    // gather endpoint features (float4 loads) -> f16 edge_in tile [32][672]
    {
        const float4* hv = reinterpret_cast<const float4*>(h);
        for (int idx = tid; idx < 32 * 80; idx += 256) {
            int e = idx / 80, q = idx - e * 80;          // q: float4 index within row
            float4 a = hv[s_row[e] * 80 + q];
            float4 b = hv[s_col[e] * 80 + q];
            v4h ah = { (_Float16)a.x, (_Float16)a.y, (_Float16)a.z, (_Float16)a.w };
            v4h bh = { (_Float16)b.x, (_Float16)b.y, (_Float16)b.z, (_Float16)b.w };
            *reinterpret_cast<v4h*>(&s_ein[e * K_E1 + q * 4])      = ah;
            *reinterpret_cast<v4h*>(&s_ein[e * K_E1 + NF + q * 4]) = bh;
        }
    }
    __syncthreads();

    // GEMM1: [32x672] x We1^T -> SiLU -> s_m1
    for (int nt = wave; nt < HID / 16; nt += 8) {
        int n = nt * 16 + (lane & 15);
        v8f acc0, acc1;
        wmma_2m(s_ein, K_E1, We1, K_E1, K_E1, nt, be1[n], lane, acc0, acc1);
        int rb = (lane >> 4) * 8;
#pragma unroll
        for (int r = 0; r < 8; ++r) {
            s_m1[(rb + r) * HID + n]      = (_Float16)silu_f(acc0[r]);
            s_m1[(16 + rb + r) * HID + n] = (_Float16)silu_f(acc1[r]);
        }
    }
    __syncthreads();

    // GEMM2: -> SiLU -> s_m2, and scatter-add messages into per-node agg (L2 atomics)
    for (int nt = wave; nt < HID / 16; nt += 8) {
        int n = nt * 16 + (lane & 15);
        v8f acc0, acc1;
        wmma_2m(s_m1, HID, We2, HID, HID, nt, be2[n], lane, acc0, acc1);
        int rb = (lane >> 4) * 8;
#pragma unroll
        for (int r = 0; r < 8; ++r) {
            float y0 = silu_f(acc0[r]);
            float y1 = silu_f(acc1[r]);
            s_m2[(rb + r) * HID + n]      = (_Float16)y0;
            s_m2[(16 + rb + r) * HID + n] = (_Float16)y1;
            atomicAdd(&agg[s_row[rb + r] * HID + n], y0);
            atomicAdd(&agg[s_row[16 + rb + r] * HID + n], y1);
        }
    }
    __syncthreads();

    // GEMM3 (coord head hidden): -> SiLU -> s_m1 (reuse as c)
    for (int nt = wave; nt < HID / 16; nt += 8) {
        int n = nt * 16 + (lane & 15);
        v8f acc0, acc1;
        wmma_2m(s_m2, HID, Wc1, HID, HID, nt, bc1[n], lane, acc0, acc1);
        int rb = (lane >> 4) * 8;
#pragma unroll
        for (int r = 0; r < 8; ++r) {
            s_m1[(rb + r) * HID + n]      = (_Float16)silu_f(acc0[r]);
            s_m1[(16 + rb + r) * HID + n] = (_Float16)silu_f(acc1[r]);
        }
    }
    __syncthreads();

    // per-edge scalar w = c . Wc2 ; scatter coord_diff * w
    if (tid < 32) {
        int e = tid;
        float acc = 0.0f;
        for (int k = 0; k < HID; ++k) acc += (float)s_m1[e * HID + k] * Wc2[k];
        int nd = s_row[e];
        atomicAdd(&coord_agg[nd * 3 + 0], s_cd[e * 3 + 0] * acc);
        atomicAdd(&coord_agg[nd * 3 + 1], s_cd[e * 3 + 1] * acc);
        atomicAdd(&coord_agg[nd * 3 + 2], s_cd[e * 3 + 2] * acc);
    }
}

// ---------------- node kernel: 32 nodes per block, 8 waves -----------------
__global__ __launch_bounds__(256)
void egcl_node_kernel(const float* __restrict__ h, const float* __restrict__ coord,
                      const float* __restrict__ agg, const float* __restrict__ coord_agg,
                      const _Float16* __restrict__ Wn1, const float* __restrict__ bn1,
                      const _Float16* __restrict__ Wn2, const float* __restrict__ bn2,
                      float* __restrict__ hout, float* __restrict__ coordout) {
    __shared__ __align__(16) _Float16 s_in[32 * K_N1];    // 40960 B
    __shared__ __align__(16) _Float16 s_t [32 * HID];     // 20480 B

    const int tid  = threadIdx.x;
    const int lane = tid & 31;
    const int wave = tid >> 5;
    const int n0   = blockIdx.x * 32;

    {
        const float4* hv = reinterpret_cast<const float4*>(h);
        const float4* av = reinterpret_cast<const float4*>(agg);
        for (int idx = tid; idx < 32 * 80; idx += 256) {
            int e = idx / 80, q = idx - e * 80;
            int nd = n0 + e; if (nd >= NODES) nd = NODES - 1;   // clamp (garbage rows unguarded in LDS only)
            float4 a = hv[nd * 80 + q];
            float4 b = av[nd * 80 + q];
            v4h ah = { (_Float16)a.x, (_Float16)a.y, (_Float16)a.z, (_Float16)a.w };
            v4h bh = { (_Float16)b.x, (_Float16)b.y, (_Float16)b.z, (_Float16)b.w };
            *reinterpret_cast<v4h*>(&s_in[e * K_N1 + q * 4])      = ah;
            *reinterpret_cast<v4h*>(&s_in[e * K_N1 + NF + q * 4]) = bh;
        }
    }
    __syncthreads();

    // GEMM1: [32x640] x Wn1^T -> SiLU -> s_t
    for (int nt = wave; nt < HID / 16; nt += 8) {
        int n = nt * 16 + (lane & 15);
        v8f acc0, acc1;
        wmma_2m(s_in, K_N1, Wn1, K_N1, K_N1, nt, bn1[n], lane, acc0, acc1);
        int rb = (lane >> 4) * 8;
#pragma unroll
        for (int r = 0; r < 8; ++r) {
            s_t[(rb + r) * HID + n]      = (_Float16)silu_f(acc0[r]);
            s_t[(16 + rb + r) * HID + n] = (_Float16)silu_f(acc1[r]);
        }
    }
    __syncthreads();

    // GEMM2: -> + residual h -> h_out (guarded tail)
    for (int nt = wave; nt < NF / 16; nt += 8) {
        int n = nt * 16 + (lane & 15);
        v8f acc0, acc1;
        wmma_2m(s_t, HID, Wn2, HID, HID, nt, bn2[n], lane, acc0, acc1);
        int rb = (lane >> 4) * 8;
#pragma unroll
        for (int r = 0; r < 8; ++r) {
            int m0 = n0 + rb + r;
            int m1 = n0 + 16 + rb + r;
            if (m0 < NODES) hout[m0 * NF + n] = h[m0 * NF + n] + acc0[r];
            if (m1 < NODES) hout[m1 * NF + n] = h[m1 * NF + n] + acc1[r];
        }
    }

    if (tid < 96) {
        int e = tid / 3, j = tid - e * 3;
        int nd = n0 + e;
        if (nd < NODES)
            coordout[nd * 3 + j] = coord[nd * 3 + j] + coord_agg[nd * 3 + j];
    }
}

extern "C" void kernel_launch(void* const* d_in, const int* in_sizes, int n_in,
                              void* d_out, int out_size, void* d_ws, size_t ws_size,
                              hipStream_t stream) {
    (void)in_sizes; (void)n_in; (void)out_size; (void)ws_size;
    const float* h     = (const float*)d_in[0];
    const float* coord = (const float*)d_in[1];
    const int*   eidx  = (const int*)d_in[2];
    const float* We1f = (const float*)d_in[3];  const float* be1 = (const float*)d_in[4];
    const float* We2f = (const float*)d_in[5];  const float* be2 = (const float*)d_in[6];
    const float* Wn1f = (const float*)d_in[7];  const float* bn1 = (const float*)d_in[8];
    const float* Wn2f = (const float*)d_in[9];  const float* bn2 = (const float*)d_in[10];
    const float* Wc1f = (const float*)d_in[11]; const float* bc1 = (const float*)d_in[12];
    const float* Wc2  = (const float*)d_in[13];

    char* ws = (char*)d_ws;                           // offsets all 64B-aligned
    _Float16* We1h = (_Float16*)(ws + 0);             // 320*672*2 = 430080
    _Float16* We2h = (_Float16*)(ws + 430080);        // 320*320*2 = 204800
    _Float16* Wc1h = (_Float16*)(ws + 634880);        // 204800
    _Float16* Wn1h = (_Float16*)(ws + 839680);        // 320*640*2 = 409600
    _Float16* Wn2h = (_Float16*)(ws + 1249280);       // 204800
    float* agg       = (float*)(ws + 1454080);        // N*320*4 = 64,000,000
    float* coord_agg = (float*)(ws + 65454080);       // N*3*4   = 600,000 (contiguous after agg)

    auto cvt = [&](const float* s, _Float16* d, int rows, int cols, int ldk) {
        int tot = rows * ldk;
        cvt_w_kernel<<<(tot + 255) / 256, 256, 0, stream>>>(s, d, rows, cols, ldk);
    };
    cvt(We1f, We1h, 320, 641, K_E1);
    cvt(We2f, We2h, 320, 320, 320);
    cvt(Wc1f, Wc1h, 320, 320, 320);
    cvt(Wn1f, Wn1h, 320, 640, K_N1);
    cvt(Wn2f, Wn2h, 320, 320, 320);

    int zn = NODES * HID + NODES * 3;                 // agg + coord_agg contiguous
    zero_kernel<<<(zn + 255) / 256, 256, 0, stream>>>(agg, zn);

    egcl_edge_kernel<<<EDGES / 32, 256, 0, stream>>>(h, coord, eidx,
                                                     We1h, be1, We2h, be2,
                                                     Wc1h, bc1, Wc2,
                                                     agg, coord_agg);

    float* hout = (float*)d_out;
    float* coordout = hout + NODES * NF;
    egcl_node_kernel<<<(NODES + 31) / 32, 256, 0, stream>>>(h, coord, agg, coord_agg,
                                                            Wn1h, bn1, Wn2h, bn2,
                                                            hout, coordout);
}
